// HTSA_63763084477088
// MI455X (gfx1250) — matrix operand order repaired and checked by reading, not compiled, not run on hardware
//
#include <hip/hip_runtime.h>
#include <hip/hip_bf16.h>

typedef __attribute__((ext_vector_type(16))) _Float16 v16h;
typedef __attribute__((ext_vector_type(8)))  float    v8f;
typedef __attribute__((ext_vector_type(4)))  unsigned int u32x4;
typedef __attribute__((ext_vector_type(8)))  int      i32x8;
typedef __attribute__((ext_vector_type(4)))  int      i32x4;

#define T_LEN 2016
#define N_SER 8
#define H_NUM 4
#define HD    16
#define NH    32
#define TOK   16128
#define SCALE 0.25f

#if defined(__has_builtin)
#if __has_builtin(__builtin_amdgcn_tensor_load_to_lds) && \
    __has_builtin(__builtin_amdgcn_s_wait_tensorcnt)
#define USE_TDM 1
#endif
#endif
#ifndef USE_TDM
#define USE_TDM 0
#endif

// --- CDNA5 WMMA f16 fragment index maps (ISA 7.12.2, wave32) ---
__device__ __forceinline__ int amapK(int e, int lane) {
    return (e & 7) + 8 * (((e >> 3) << 1) + (lane >> 4));
}
__device__ __forceinline__ int bmapK(int e, int lane) {
    return (e & 7) + 8 * (((e >> 3) << 1) + (lane >> 4));
}
// C/D: VGPR r, lane -> row M = r + 8*(lane>>4), col N = lane&15.

__device__ __forceinline__ v8f wmma_f16(v16h a, v16h b, v8f c) {
    return __builtin_amdgcn_wmma_f32_16x16x32_f16(false, a, false, b,
                                                  (short)0, c, false, false);
}

// ---- Tensor Data Mover: 2D tile Global -> LDS (ISA ch.8 D# layout) --------
#if USE_TDM
__device__ __forceinline__ void tdm_tile_2d(unsigned lds_off, const void* g,
                                            unsigned dsz_code,  // 0:1B 1:2B 2:4B 3:8B
                                            unsigned tile_w, unsigned tile_h,
                                            unsigned rstride /*elements*/) {
    unsigned long long ga = (unsigned long long)(uintptr_t)g;
    u32x4 g0;
    g0[0] = 1u;                                   // count=1, user descriptor
    g0[1] = lds_off;                              // lds_addr (bytes)
    g0[2] = (unsigned)ga;                         // global_addr[31:0]
    g0[3] = (unsigned)((ga >> 32) & 0x01FFFFFFu)  // global_addr[56:32]
            | (2u << 30);                         // type = 2 ("image")
    i32x8 g1;
    g1[0] = (int)(dsz_code << 16);                // wg_mask=0, data_size
    g1[1] = (int)((tile_w & 0xFFFFu) << 16);      // tensor_dim0 = tile_w
    g1[2] = (int)(((unsigned)tile_h & 0xFFFFu) << 16); // dim0 hi=0, tensor_dim1=tile_h
    g1[3] = (int)((tile_w & 0xFFFFu) << 16);      // dim1 hi=0, tile_dim0 = tile_w
    g1[4] = (int)(tile_h & 0xFFFFu);              // tile_dim1 = tile_h, tile_dim2=0
    g1[5] = (int)rstride;                         // tensor_dim0_stride[31:0]
    g1[6] = 0;                                    // stride hi, dim1_stride lo
    g1[7] = 0;
    i32x4 z4 = {0, 0, 0, 0};                      // groups 2/3 unused (2D)
    i32x8 z8 = {0, 0, 0, 0, 0, 0, 0, 0};          // trailing group (clang-23 form)
    __builtin_amdgcn_tensor_load_to_lds(g0, g1, z4, z4, z8, 0);
}
#endif

__device__ __forceinline__ unsigned lds_off_of(const void* p) {
    return (unsigned)(uintptr_t)p;   // low 32 bits of generic ptr = LDS offset
}

__device__ __forceinline__ void stage_f16(_Float16* lds, const _Float16* g,
                                          int tw, int th, int rstride, int lane) {
#if USE_TDM
    tdm_tile_2d(lds_off_of(lds), g, 1u, (unsigned)tw, (unsigned)th,
                (unsigned)rstride);
#else
    for (int idx = lane; idx < tw * th; idx += 32)
        lds[idx] = g[(idx / tw) * rstride + (idx % tw)];
#endif
}

__device__ __forceinline__ void stage_f32(float* lds, const float* g,
                                          int tw, int th, int rstride, int lane) {
#if USE_TDM
    tdm_tile_2d(lds_off_of(lds), g, 2u, (unsigned)tw, (unsigned)th,
                (unsigned)rstride);
#else
    for (int idx = lane; idx < tw * th; idx += 32)
        lds[idx] = g[(idx / tw) * rstride + (idx % tw)];
#endif
}

__device__ __forceinline__ void stage_wait() {
#if USE_TDM
    __builtin_amdgcn_s_wait_tensorcnt(0);
    asm volatile("" ::: "memory");
#endif
    __syncthreads();
}

// -------------------------------------------------------------------------
// Kernel 1: q/k/v = x @ W^T. TDM stages the x tile and W tile into LDS.
// Also writes weekly third of concat buffer (weekly attention == identity).
// grid: (1008, 12) y = w*4 + head ; block: 32
// -------------------------------------------------------------------------
__global__ void HTSA_qkv_proj(const float* __restrict__ x,
                              const float* __restrict__ Wq,
                              const float* __restrict__ Wk,
                              const float* __restrict__ Wv,
                              _Float16* __restrict__ qh,
                              _Float16* __restrict__ kh,
                              _Float16* __restrict__ vh,
                              _Float16* __restrict__ oc) {
    __shared__ float lx[16 * 64];
    __shared__ float lw[16 * 64];
    const int lane = threadIdx.x;
    const int tile = blockIdx.x;
    const int w    = blockIdx.y >> 2;
    const int h    = blockIdx.y & 3;
    const float* __restrict__ W = (w == 0) ? Wq : (w == 1) ? Wk : Wv;
    _Float16* __restrict__ dst  = (w == 0) ? qh : (w == 1) ? kh : vh;

    stage_f32(lx, x + (size_t)tile * 16 * 64, 64, 16, 64, lane);
    stage_f32(lw, W + (size_t)h * 16 * 64,    64, 16, 64, lane);
    stage_wait();

    v8f c = {};
#pragma unroll
    for (int kk = 0; kk < 64; kk += 32) {
        v16h a, b;
#pragma unroll
        for (int e = 0; e < 16; ++e) {
            a[e] = (_Float16)lx[(lane & 15) * 64 + kk + amapK(e, lane)];
            b[e] = (_Float16)lw[(lane & 15) * 64 + kk + bmapK(e, lane)];
        }
        c = wmma_f16(a, b, c);
    }
#pragma unroll
    for (int r = 0; r < 8; ++r) {
        const int Mo = r + 8 * (lane >> 4);
        const int go = tile * 16 + Mo;
        const int t  = go >> 3, n = go & 7;
        const int dh = lane & 15;
        const _Float16 hv = (_Float16)c[r];
        dst[(((n * H_NUM + h) * T_LEN) + t) * HD + dh] = hv;
        if (w == 2)
            oc[(size_t)go * 192 + 128 + h * 16 + dh] = hv;
    }
}

// -------------------------------------------------------------------------
// Kernel 2: local attention (<=12 trailing keys). 32-key window staged by TDM.
// grid: (126, 32) ; block: 32
// -------------------------------------------------------------------------
__global__ void HTSA_local_attn(const _Float16* __restrict__ qh,
                                const _Float16* __restrict__ kh,
                                const _Float16* __restrict__ vh,
                                _Float16* __restrict__ oc) {
    __shared__ _Float16 lq[16 * 16];
    __shared__ _Float16 lk[32 * 16];
    __shared__ _Float16 lv[32 * 16];
    __shared__ _Float16 lp[16 * 32];
    const int lane = threadIdx.x;
    const int qt = blockIdx.x;
    const int nh = blockIdx.y;
    const int h = nh & 3, n = nh >> 2;
    const _Float16* __restrict__ Q = qh + (size_t)nh * T_LEN * HD;
    const _Float16* __restrict__ K = kh + (size_t)nh * T_LEN * HD;
    const _Float16* __restrict__ V = vh + (size_t)nh * T_LEN * HD;
    const int jb = (qt == 0) ? 0 : qt * 16 - 16;   // window base, always in-range

    stage_f16(lq, Q + (size_t)qt * 16 * HD, 16, 16, 16, lane);
    stage_f16(lk, K + (size_t)jb * HD,      16, 32, 16, lane);
    stage_f16(lv, V + (size_t)jb * HD,      16, 32, 16, lane);
    stage_wait();

    v16h aq;
#pragma unroll
    for (int e = 0; e < 16; ++e) {
        const int k = amapK(e, lane);
        aq[e] = (k < HD) ? lq[(lane & 15) * 16 + k] : (_Float16)0.0f;
    }

    float s[2][8];
#pragma unroll
    for (int g = 0; g < 2; ++g) {
        v16h b;
#pragma unroll
        for (int e = 0; e < 16; ++e) {
            const int k = bmapK(e, lane);
            b[e] = (k < HD) ? lk[(g * 16 + (lane & 15)) * 16 + k] : (_Float16)0.0f;
        }
        v8f cs = {};
        cs = wmma_f16(aq, b, cs);
#pragma unroll
        for (int r = 0; r < 8; ++r) s[g][r] = cs[r] * SCALE;
    }

#pragma unroll
    for (int r = 0; r < 8; ++r) {
        const int Mi = r + 8 * (lane >> 4);
        const int i  = qt * 16 + Mi;
#pragma unroll
        for (int g = 0; g < 2; ++g) {
            const int j = jb + g * 16 + (lane & 15);
            const bool ok = (j <= i) && (i - j < 12);
            s[g][r] = ok ? s[g][r] : -1e30f;
        }
        float m = fmaxf(s[0][r], s[1][r]);
#pragma unroll
        for (int d = 1; d < 16; d <<= 1) m = fmaxf(m, __shfl_xor(m, d, 32));
        const float p0 = __expf(s[0][r] - m);
        const float p1 = __expf(s[1][r] - m);
        float sum = p0 + p1;
#pragma unroll
        for (int d = 1; d < 16; d <<= 1) sum += __shfl_xor(sum, d, 32);
        const float inv = 1.0f / sum;
        lp[Mi * 32 +      (lane & 15)] = (_Float16)(p0 * inv);
        lp[Mi * 32 + 16 + (lane & 15)] = (_Float16)(p1 * inv);
    }
    __syncthreads();

    v16h ap, bv;
#pragma unroll
    for (int e = 0; e < 16; ++e) {
        ap[e] = lp[(lane & 15) * 32 + amapK(e, lane)];
        bv[e] = lv[bmapK(e, lane) * 16 + (lane & 15)];
    }
    v8f co = {};
    co = wmma_f16(ap, bv, co);
#pragma unroll
    for (int r = 0; r < 8; ++r) {
        const int Mi = r + 8 * (lane >> 4);
        const int t  = qt * 16 + Mi;
        const int go = t * N_SER + n;
        oc[(size_t)go * 192 + h * 16 + (lane & 15)] = (_Float16)co[r];
    }
}

// -------------------------------------------------------------------------
// Kernel 3: daily attention: causal 7x7 per residue class rho (t = rho+288m),
// padded to one 16x16 WMMA tile. TDM does the strided 7-row gathers.
// grid: (288, 32) ; block: 32
// -------------------------------------------------------------------------
__global__ void HTSA_daily_attn(const _Float16* __restrict__ qh,
                                const _Float16* __restrict__ kh,
                                const _Float16* __restrict__ vh,
                                _Float16* __restrict__ oc) {
    __shared__ _Float16 lq[16 * 16];
    __shared__ _Float16 lk[16 * 16];
    __shared__ _Float16 lv[16 * 16];
    __shared__ _Float16 lp[16 * 32];
    const int lane = threadIdx.x;
    const int rho = blockIdx.x;
    const int nh  = blockIdx.y;
    const int h = nh & 3, n = nh >> 2;
    const _Float16* __restrict__ Q = qh + (size_t)nh * T_LEN * HD;
    const _Float16* __restrict__ K = kh + (size_t)nh * T_LEN * HD;
    const _Float16* __restrict__ V = vh + (size_t)nh * T_LEN * HD;

    // zero pad rows 7..15 (disjoint from the TDM's rows 0..6 -> no ordering hazard)
    for (int idx = lane; idx < 9 * 16; idx += 32) {
        lq[7 * 16 + idx] = (_Float16)0.0f;
        lk[7 * 16 + idx] = (_Float16)0.0f;
        lv[7 * 16 + idx] = (_Float16)0.0f;
    }
    stage_f16(lq, Q + (size_t)rho * HD, 16, 7, 288 * HD, lane);
    stage_f16(lk, K + (size_t)rho * HD, 16, 7, 288 * HD, lane);
    stage_f16(lv, V + (size_t)rho * HD, 16, 7, 288 * HD, lane);
    stage_wait();

    v16h aq, bk;
#pragma unroll
    for (int e = 0; e < 16; ++e) {
        const int ka = amapK(e, lane);
        aq[e] = (ka < HD) ? lq[(lane & 15) * 16 + ka] : (_Float16)0.0f;
        const int kb = bmapK(e, lane);
        bk[e] = (kb < HD) ? lk[(lane & 15) * 16 + kb] : (_Float16)0.0f;
    }
    v8f cs = {};
    cs = wmma_f16(aq, bk, cs);

#pragma unroll
    for (int r = 0; r < 8; ++r) {
        const int mq = r + 8 * (lane >> 4);
        const int mk = lane & 15;
        const bool ok = (mq < 7) && (mk <= mq);
        float val = ok ? cs[r] * SCALE : -1e30f;
        float m = val;
#pragma unroll
        for (int d = 1; d < 16; d <<= 1) m = fmaxf(m, __shfl_xor(m, d, 32));
        float p = __expf(val - m);
        float sum = p;
#pragma unroll
        for (int d = 1; d < 16; d <<= 1) sum += __shfl_xor(sum, d, 32);
        p = p / sum;
        lp[mq * 32 +      mk] = (_Float16)p;
        lp[mq * 32 + 16 + mk] = (_Float16)0.0f;
    }
    __syncthreads();

    v16h ap, bv;
#pragma unroll
    for (int e = 0; e < 16; ++e) {
        ap[e] = lp[(lane & 15) * 32 + amapK(e, lane)];
        const int kr = bmapK(e, lane);
        bv[e] = (kr < HD) ? lv[kr * 16 + (lane & 15)] : (_Float16)0.0f;
    }
    v8f co = {};
    co = wmma_f16(ap, bv, co);
#pragma unroll
    for (int r = 0; r < 8; ++r) {
        const int mq = r + 8 * (lane >> 4);
        if (mq < 7) {
            const int t  = rho + 288 * mq;
            const int go = t * N_SER + n;
            oc[(size_t)go * 192 + 64 + h * 16 + (lane & 15)] = (_Float16)co[r];
        }
    }
}

// -------------------------------------------------------------------------
// Kernel 4: out = o(16128x192) @ Wp^T + bp. Contiguous 16-row panels via TDM.
// grid: (1008, 4) ; block: 32
// -------------------------------------------------------------------------
__global__ void HTSA_out_proj(const _Float16* __restrict__ oc,
                              const float* __restrict__ Wp,
                              const float* __restrict__ bp,
                              float* __restrict__ out) {
    __shared__ _Float16 lo[16 * 192];
    __shared__ float    lwp[16 * 192];
    const int lane = threadIdx.x;
    const int tile = blockIdx.x;
    const int d0   = blockIdx.y * 16;

    stage_f16(lo,  oc + (size_t)tile * 16 * 192, 192 * 16, 1, 192 * 16, lane);
    stage_f32(lwp, Wp + (size_t)d0 * 192,        192 * 16, 1, 192 * 16, lane);
    stage_wait();

    v8f c = {};
#pragma unroll
    for (int kk = 0; kk < 192; kk += 32) {
        v16h a, b;
#pragma unroll
        for (int e = 0; e < 16; ++e) {
            a[e] = lo[(lane & 15) * 192 + kk + amapK(e, lane)];
            b[e] = (_Float16)lwp[(lane & 15) * 192 + kk + bmapK(e, lane)];
        }
        c = wmma_f16(a, b, c);
    }
#pragma unroll
    for (int r = 0; r < 8; ++r) {
        const int Mo = r + 8 * (lane >> 4);
        const int go = tile * 16 + Mo;
        out[(size_t)go * 64 + d0 + (lane & 15)] = c[r] + bp[d0 + (lane & 15)];
    }
}

extern "C" void kernel_launch(void* const* d_in, const int* in_sizes, int n_in,
                              void* d_out, int out_size, void* d_ws, size_t ws_size,
                              hipStream_t stream) {
    (void)in_sizes; (void)n_in; (void)out_size; (void)ws_size;
    const float* x  = (const float*)d_in[0];
    const float* Wq = (const float*)d_in[1];
    const float* Wk = (const float*)d_in[2];
    const float* Wv = (const float*)d_in[3];
    const float* Wp = (const float*)d_in[4];
    const float* bp = (const float*)d_in[5];
    float* out = (float*)d_out;

    char* ws = (char*)d_ws;
    const size_t qkv_bytes = (size_t)NH * T_LEN * HD * sizeof(_Float16);
    _Float16* qh = (_Float16*)(ws);
    _Float16* kh = (_Float16*)(ws + qkv_bytes);
    _Float16* vh = (_Float16*)(ws + 2 * qkv_bytes);
    _Float16* oc = (_Float16*)(ws + 3 * qkv_bytes);

    const dim3 b32(32, 1, 1);
    HTSA_qkv_proj <<<dim3(1008, 12, 1), b32, 0, stream>>>(x, Wq, Wk, Wv, qh, kh, vh, oc);
    HTSA_local_attn<<<dim3(126, 32, 1), b32, 0, stream>>>(qh, kh, vh, oc);
    HTSA_daily_attn<<<dim3(288, 32, 1), b32, 0, stream>>>(qh, kh, vh, oc);
    HTSA_out_proj <<<dim3(1008, 4, 1), b32, 0, stream>>>(oc, Wp, bp, out);
}
